// CLinear_81990925680772
// MI455X (gfx1250) — matrix operand order repaired and checked by reading, not compiled, not run on hardware
//
#include <hip/hip_runtime.h>

// ---- types -----------------------------------------------------------------
typedef _Float16 v2h  __attribute__((ext_vector_type(2)));
typedef _Float16 v8h  __attribute__((ext_vector_type(8)));
typedef _Float16 v16h __attribute__((ext_vector_type(16)));
typedef float    v8f  __attribute__((ext_vector_type(8)));
typedef float    v4f  __attribute__((ext_vector_type(4)));
typedef int      v4i  __attribute__((ext_vector_type(4)));

// ---- problem constants -----------------------------------------------------
#define K_DIM   4096
#define N_DIM   4096
#define GROUPS  64          // quant groups along K, group size 64
#define BM      128         // block tile M
#define BN      128         // block tile N
#define BK      64          // K tile == quant group size
#define NK      (K_DIM / BK)
#define LDSTR   72          // padded LDS row stride in f16 (64 + 8)

__global__ __launch_bounds__(256)
void qlinear_wmma_kernel(const float* __restrict__ x,
                         const int*   __restrict__ packed,
                         const float* __restrict__ mn,
                         const float* __restrict__ scale,
                         const float* __restrict__ bias,
                         float*       __restrict__ out)
{
    // ping-pong LDS: 2 x (18 KB + 18 KB) = 72 KB of the 320 KB WGP pool
    __shared__ _Float16 Xs[2][BM * LDSTR];
    __shared__ _Float16 Ws[2][BN * LDSTR];

    const int tid  = threadIdx.x;
    const int lane = tid & 31;
    const int wave = tid >> 5;

    const int m_blk = blockIdx.y * BM;
    const int n_blk = blockIdx.x * BN;

    // wave sub-tile: 32 (M) x 64 (N); waves arranged 4 (M) x 2 (N)
    const int wm = (wave & 3) * 32;
    const int wn = (wave >> 2) * 64;

    const int hi  = lane >> 4;    // lane half: 0 or 1
    const int l16 = lane & 15;

    v8f acc[2][4];
#pragma unroll
    for (int i = 0; i < 2; ++i)
#pragma unroll
        for (int j = 0; j < 4; ++j)
            acc[i][j] = v8f{};

    // X staging: per pass 16 rows x 16 float4 (covers 128x64 f32 in 8 passes)
    const int xcol4 = tid & 15;
    const int xrow  = tid >> 4;
    // W staging: per pass 32 rows x 8 int4 (covers 128x32 ints in 4 passes)
    const int wi4   = tid & 7;
    const int wrow  = tid >> 3;

    // ---- register pipeline stage -------------------------------------------
    v4f   xr[8];
    v4i   wr[4];
    float sv[4], mv[4];

    auto load_tiles = [&](int kt) {
        const int k0 = kt * BK;
#pragma unroll
        for (int p = 0; p < 8; ++p) {
            const int r = xrow + p * 16;
            xr[p] = *(const v4f*)(x + (size_t)(m_blk + r) * K_DIM + k0 + xcol4 * 4);
        }
#pragma unroll
        for (int p = 0; p < 4; ++p) {
            const int n  = wrow + p * 32;
            const int ng = (n_blk + n) * GROUPS + kt;   // group g == kt (BK == 64)
            wr[p] = *(const v4i*)(packed + (size_t)ng * 32 + wi4 * 4);
            sv[p] = scale[ng];
            mv[p] = mn[ng];
        }
    };

    auto store_tiles = [&](int buf) {
#pragma unroll
        for (int p = 0; p < 8; ++p) {
            const int r = xrow + p * 16;
            _Float16* dst = &Xs[buf][r * LDSTR + xcol4 * 4];
            v2h a = { (_Float16)xr[p].x, (_Float16)xr[p].y };
            v2h b = { (_Float16)xr[p].z, (_Float16)xr[p].w };
            *(v2h*)(dst)     = a;
            *(v2h*)(dst + 2) = b;
        }
        // packed-f16 dequant: f16 bits (0x6400 | v) == 1024 + v exactly.
        // Subtract 1024 exactly (Sterbenz) then one pk_fma per value pair.
#pragma unroll
        for (int p = 0; p < 4; ++p) {
            const int n = wrow + p * 32;
            const float invf = 1.0f / sv[p];            // reference: val/scale + mn
            const _Float16 invh = (_Float16)invf;
            const _Float16 mnh  = (_Float16)mv[p];
            const v2h inv2  = { invh, invh };
            const v2h mn2   = { mnh,  mnh  };
            const v2h k1024 = { (_Float16)1024.0f, (_Float16)1024.0f };
            _Float16* dh = &Ws[buf][n * LDSTR + wi4 * 4]; // hi nibble: first half of group
            _Float16* dl = dh + 32;                       // lo nibble: second half
#pragma unroll
            for (int bp = 0; bp < 2; ++bp) {
                const unsigned c = (unsigned)wr[p][2 * bp] |
                                   ((unsigned)wr[p][2 * bp + 1] << 16);
                const unsigned lo_bits = ( c        & 0x000F000Fu) | 0x64006400u;
                const unsigned hi_bits = ((c >> 4)  & 0x000F000Fu) | 0x64006400u;
                v2h hv = __builtin_bit_cast(v2h, hi_bits) - k1024;  // exact
                v2h lv = __builtin_bit_cast(v2h, lo_bits) - k1024;  // exact
                *(v2h*)(dh + 2 * bp) = hv * inv2 + mn2;             // v_pk_fma_f16
                *(v2h*)(dl + 2 * bp) = lv * inv2 + mn2;
            }
        }
    };

    // ---- prologue: stage tile 0 into registers -----------------------------
    load_tiles(0);

    for (int kt = 0; kt < NK; ++kt) {
        const int buf = kt & 1;

        // drain register stage into this iteration's LDS buffer
        store_tiles(buf);
        __syncthreads();   // single barrier per iteration (ping-pong makes it safe)

        // prefetch next tile's globals (uniform clamp — no branch in hot loop);
        // latency hidden behind the WMMA phase
        load_tiles(kt + 1 < NK ? kt + 1 : kt);

        // ---- WMMA phase: 2 x (K=32) sub-steps on buffer `buf` --------------
#pragma unroll
        for (int ks = 0; ks < 2; ++ks) {
            const int kb = ks * 32;

            // A fragments (16-bit 16x32): lanes 0-15 K=0..7 & 16..23,
            // lanes 16-31 K=8..15 & 24..31 (two contiguous 8-f16 runs)
            v16h afrag[2];
#pragma unroll
            for (int ms = 0; ms < 2; ++ms) {
                const _Float16* src = &Xs[buf][(wm + ms * 16 + l16) * LDSTR + kb + 8 * hi];
                *((v8h*)&afrag[ms])     = *(const v8h*)src;
                *((v8h*)&afrag[ms] + 1) = *(const v8h*)(src + 16);
            }

            // B fragments (32x16): lanes 0-15 K=0..7 & 16..23, lanes 16-31 K=8..15 & 24..31;
            // two contiguous 8-f16 runs per lane, N = lane & 15
            v16h bfrag[4];
#pragma unroll
            for (int ns = 0; ns < 4; ++ns) {
                const _Float16* src = &Ws[buf][(wn + ns * 16 + l16) * LDSTR + kb + 8 * hi];
                *((v8h*)&bfrag[ns])     = *(const v8h*)src;
                *((v8h*)&bfrag[ns] + 1) = *(const v8h*)(src + 16);
            }

#pragma unroll
            for (int ms = 0; ms < 2; ++ms)
#pragma unroll
                for (int ns = 0; ns < 4; ++ns)
                    acc[ms][ns] = __builtin_amdgcn_wmma_f32_16x16x32_f16(
                        /*neg_a=*/false, afrag[ms],
                        /*neg_b=*/false, bfrag[ns],
                        /*c_mod=*/(short)0, acc[ms][ns],
                        /*reuse_a=*/false, /*reuse_b=*/false);
        }
    }

    // ---- epilogue: D layout — VGPR r -> M = r + 8*hi, N = lane&15 ----------
    // Output is write-once: non-temporal stores keep x resident in the 192MB L2.
#pragma unroll
    for (int ns = 0; ns < 4; ++ns) {
        const int col = n_blk + wn + ns * 16 + l16;
        const float bv = bias[col];
#pragma unroll
        for (int ms = 0; ms < 2; ++ms) {
#pragma unroll
            for (int r = 0; r < 8; ++r) {
                const int row = m_blk + wm + ms * 16 + r + 8 * hi;
                __builtin_nontemporal_store(acc[ms][ns][r] + bv,
                                            out + (size_t)row * N_DIM + col);
            }
        }
    }
}

extern "C" void kernel_launch(void* const* d_in, const int* in_sizes, int n_in,
                              void* d_out, int out_size, void* d_ws, size_t ws_size,
                              hipStream_t stream) {
    const float* x      = (const float*)d_in[0];
    const int*   packed = (const int*)  d_in[1];
    const float* mn     = (const float*)d_in[2];
    const float* scale  = (const float*)d_in[3];
    const float* bias   = (const float*)d_in[4];
    float*       out    = (float*)d_out;

    const int M = in_sizes[0] / K_DIM;          // 8192 = 4 * 2048
    dim3 grid(N_DIM / BN, M / BM);              // (32, 64)
    qlinear_wmma_kernel<<<grid, dim3(256), 0, stream>>>(x, packed, mn, scale, bias, out);
}
